// MultiStreamLocalExperts_5566277615660
// MI455X (gfx1250) — matrix operand order repaired and checked by reading, not compiled, not run on hardware
//
#include <hip/hip_runtime.h>

// ---------------------------------------------------------------------------
// MoE grouped expert MLP:  y = relu(x @ W1) @ W2   per expert
//   E=8, C=4096 (M), D=1024, F=4096  -> ~550 GFLOP, AI ~1000 FLOP/B:
//   compute-bound -> v_wmma_f32_16x16x32_bf16 with fp32 accumulation.
//
// Prep pass (negligible traffic vs compute):
//   Xb  = bf16(x)                      [E*C, D]
//   W1t = bf16(W1^T) per expert        [E, F, D]   (K-contiguous)
//   W2t = bf16(W2^T) per expert        [E, D, F]   (K-contiguous)
// GEMM1: Xb  x W1t -> relu -> H (bf16) [E, C, F]
// GEMM2: H   x W2t -> Y (fp32)         [E*C, D]
//
// GEMM: 256 thr = 8 wave32, block tile 128Mx256N, wave tile 64x64
//       (16 accum frags), K-step 32, double-buffered LDS (60 KB),
//       async global->LDS staging (global_load_async_to_lds_b128 / ASYNCcnt),
//       six running per-thread source pointers, unroll-1 K loop to stay
//       under 256 VGPRs (avoids v_mov shuffling / vgpr-msb toggles).
// ---------------------------------------------------------------------------

typedef __attribute__((ext_vector_type(16))) __bf16 v16bf;
typedef __attribute__((ext_vector_type(8)))  __bf16 v8bf;
typedef __attribute__((ext_vector_type(8)))  float  v8f;
typedef __attribute__((ext_vector_type(4)))  int    v4i;

#define TILE_M 128
#define TILE_N 256
#define TILE_K 32
#define LDSK   40                      // bf16 per LDS row: 32 data + 8 pad (80B)
#define ABUF   (TILE_M * LDSK)         // 5120 bf16
#define BBUF   (TILE_N * LDSK)         // 10240 bf16
#define BUFSZ  (ABUF + BBUF)           // 15360 bf16 = 30720 B per buffer

#ifdef __has_builtin
#if __has_builtin(__builtin_amdgcn_global_load_async_to_lds_b128)
#define HAVE_ASYNC_LDS 1
#endif
#endif
#ifndef HAVE_ASYNC_LDS
#define HAVE_ASYNC_LDS 0
#endif

union Frag16 { v16bf v; v8bf h[2]; };

__device__ __forceinline__ unsigned pk_bf16(float a, float b) {
  unsigned short lo = __builtin_bit_cast(unsigned short, (__bf16)a);
  unsigned short hi = __builtin_bit_cast(unsigned short, (__bf16)b);
  return ((unsigned)hi << 16) | (unsigned)lo;
}

__device__ __forceinline__ void wait_async0() {
#if HAVE_ASYNC_LDS
#ifdef __has_builtin
#if __has_builtin(__builtin_amdgcn_s_wait_asynccnt)
  __builtin_amdgcn_s_wait_asynccnt(0);
#else
  asm volatile("s_wait_asynccnt 0x0" ::: "memory");
#endif
#endif
#endif
}

// one 16-byte global -> LDS copy (async when available)
__device__ __forceinline__ void async_cp16(const __bf16* g, __bf16* l) {
#if HAVE_ASYNC_LDS
  __builtin_amdgcn_global_load_async_to_lds_b128(
      (__attribute__((address_space(1))) v4i*)g,
      (__attribute__((address_space(3))) v4i*)l, 0, 0);
#else
  *(uint4*)l = *(const uint4*)g;
#endif
}

// ---------------------------------------------------------------------------
// Prep kernels
// ---------------------------------------------------------------------------

// fp32 -> bf16 elementwise, 8 elems/thread, n divisible by 2048
__global__ __launch_bounds__(256)
void cvt_bf16_kernel(const float* __restrict__ x, __bf16* __restrict__ xb) {
  const size_t i = ((size_t)blockIdx.x * 256 + threadIdx.x) * 8;
  const float4 a = *(const float4*)(x + i);
  const float4 b = *(const float4*)(x + i + 4);
  uint4 p;
  p.x = pk_bf16(a.x, a.y);
  p.y = pk_bf16(a.z, a.w);
  p.z = pk_bf16(b.x, b.y);
  p.w = pk_bf16(b.z, b.w);
  *(uint4*)(xb + i) = p;
}

// W [K,N] fp32 -> Wt [N,K] bf16 per expert (64x64 tiles via LDS)
__global__ __launch_bounds__(256)
void transpose_cvt_kernel(const float* __restrict__ W, __bf16* __restrict__ Wt,
                          int K, int N) {
  __shared__ __align__(16) __bf16 t[64][72];
  const int tid = threadIdx.x;
  const size_t eoff = (size_t)blockIdx.z * K * N;
  const float* Wp  = W  + eoff;
  __bf16*      Wtp = Wt + eoff;
  const int k0 = blockIdx.y * 64;
  const int n0 = blockIdx.x * 64;
#pragma unroll
  for (int q = 0; q < 4; ++q) {
    const int idx = q * 256 + tid;     // 1024 float4 chunks
    const int kr  = idx >> 4;          // 64 rows (K)
    const int c4  = idx & 15;
    const float4 v = *(const float4*)(Wp + (size_t)(k0 + kr) * N + n0 + c4 * 4);
    const int nc = c4 * 4;
    t[nc + 0][kr] = (__bf16)v.x;
    t[nc + 1][kr] = (__bf16)v.y;
    t[nc + 2][kr] = (__bf16)v.z;
    t[nc + 3][kr] = (__bf16)v.w;
  }
  __syncthreads();
#pragma unroll
  for (int q = 0; q < 2; ++q) {
    const int idx = q * 256 + tid;     // 512 chunks of 8 bf16
    const int nr  = idx >> 3;
    const int c8  = idx & 7;
    const uint4 d = *(const uint4*)&t[nr][c8 * 8];
    *(uint4*)(Wtp + (size_t)(n0 + nr) * K + k0 + c8 * 8) = d;
  }
}

// ---------------------------------------------------------------------------
// Main GEMM:  Out[M,N] (+relu) = A[M,K] x Bt[N,K]^T   (bf16 in, f32 acc)
// N, K are compile-time so all strides fold to immediates.
// ---------------------------------------------------------------------------
template<int N, int K, bool RELU, bool OUT_BF16>
__global__ __launch_bounds__(256)
void moe_gemm(const __bf16* __restrict__ Aall, const __bf16* __restrict__ Btall,
              void* __restrict__ Oall, int M)
{
  __shared__ __align__(16) __bf16 smem[2 * BUFSZ];   // 61440 B

  const int tid  = threadIdx.x;
  const int wave = tid >> 5;
  const int lane = tid & 31;
  const int half = lane >> 4;
  const int lrow = lane & 15;
  const int wm   = wave >> 2;          // 0..1 -> 64-row band
  const int wn   = wave & 3;           // 0..3 -> 64-col band

  const int e  = blockIdx.z;
  const int m0 = blockIdx.y * TILE_M;
  const int n0 = blockIdx.x * TILE_N;

  const __bf16* A  = Aall  + (size_t)e * M * K + (size_t)m0 * K;
  const __bf16* Bt = Btall + (size_t)e * N * K + (size_t)n0 * K;

  // ---- per-thread staging: 6 running source pointers, fixed LDS offsets ----
  const int srow = tid >> 2;           // 64 rows per chunk-group
  const int sc8  = (tid & 3) * 8;      // 8-bf16 column chunk
  const __bf16* pA0 = A  + (size_t)(srow      ) * K + sc8;
  const __bf16* pA1 = A  + (size_t)(srow +  64) * K + sc8;
  const __bf16* pB0 = Bt + (size_t)(srow      ) * K + sc8;
  const __bf16* pB1 = Bt + (size_t)(srow +  64) * K + sc8;
  const __bf16* pB2 = Bt + (size_t)(srow + 128) * K + sc8;
  const __bf16* pB3 = Bt + (size_t)(srow + 192) * K + sc8;
  const int lA = srow * LDSK + sc8;
  const int lB = ABUF + srow * LDSK + sc8;

  auto stage = [&](int b) {
    __bf16* s = smem + b * BUFSZ;
    async_cp16(pA0, s + lA);
    async_cp16(pA1, s + lA + 64 * LDSK);
    async_cp16(pB0, s + lB);
    async_cp16(pB1, s + lB + 64 * LDSK);
    async_cp16(pB2, s + lB + 128 * LDSK);
    async_cp16(pB3, s + lB + 192 * LDSK);
    pA0 += TILE_K; pA1 += TILE_K;
    pB0 += TILE_K; pB1 += TILE_K; pB2 += TILE_K; pB3 += TILE_K;
  };

  v8f acc[4][4];
#pragma unroll
  for (int i = 0; i < 4; ++i)
#pragma unroll
    for (int j = 0; j < 4; ++j)
      acc[i][j] = (v8f){0.f, 0.f, 0.f, 0.f, 0.f, 0.f, 0.f, 0.f};

  // prologue: stage tile k0=0 into buffer 0
  stage(0);
  wait_async0();
  __syncthreads();

  int buf = 0;
#pragma unroll 1
  for (int k0 = 0; k0 < K; k0 += TILE_K) {
    if (k0 + TILE_K < K) stage(buf ^ 1);  // async into other buffer

    // ---- fragments per ISA bf16 A 16x32 / B 32x16 VGPR layouts ----
    const __bf16* Ab = smem + buf * BUFSZ;
    const __bf16* Bb = Ab + ABUF;
    Frag16 afr[4], bfr[4];
#pragma unroll
    for (int i = 0; i < 4; ++i) {
      // lanes 0-15: K {0..7,16..23}; lanes 16-31: K {8..15,24..31}
      const __bf16* p = Ab + (wm * 64 + i * 16 + lrow) * LDSK + half * 8;
      afr[i].h[0] = *(const v8bf*)(p);
      afr[i].h[1] = *(const v8bf*)(p + 16);
    }
#pragma unroll
    for (int j = 0; j < 4; ++j) {
      // lanes 0-15: col n, K {0..7,16..23}; lanes 16-31: col n, K {8..15,24..31}
      const __bf16* p = Bb + (wn * 64 + j * 16 + lrow) * LDSK + half * 8;
      bfr[j].h[0] = *(const v8bf*)(p);
      bfr[j].h[1] = *(const v8bf*)(p + 16);
    }
#pragma unroll
    for (int i = 0; i < 4; ++i)
#pragma unroll
      for (int j = 0; j < 4; ++j)
        acc[i][j] = __builtin_amdgcn_wmma_f32_16x16x32_bf16(
            false, afr[i].v, false, bfr[j].v,
            (short)0, acc[i][j], false, false);

    wait_async0();
    __syncthreads();
    buf ^= 1;
  }

  // ---- epilogue: C frag VGPR r -> row (half*8 + r), col lrow ----
#pragma unroll
  for (int i = 0; i < 4; ++i) {
#pragma unroll
    for (int j = 0; j < 4; ++j) {
      const int gm = m0 + wm * 64 + i * 16 + half * 8;
      const int gn = n0 + wn * 64 + j * 16 + lrow;
#pragma unroll
      for (int r = 0; r < 8; ++r) {
        float v = acc[i][j][r];
        if (RELU) v = fmaxf(v, 0.0f);
        const size_t off = (size_t)e * M * N + (size_t)(gm + r) * N + gn;
        if (OUT_BF16) ((__bf16*)Oall)[off] = (__bf16)v;
        else          ((float*)Oall)[off]  = v;
      }
    }
  }
}

// ---------------------------------------------------------------------------

extern "C" void kernel_launch(void* const* d_in, const int* in_sizes, int n_in,
                              void* d_out, int out_size, void* d_ws, size_t ws_size,
                              hipStream_t stream) {
  (void)in_sizes; (void)n_in; (void)out_size;

  constexpr int E = 8, C = 4096, D = 1024, F = 4096;

  const float* X  = (const float*)d_in[0];   // [E*C, D]
  // d_in[1]: experts_capacity_usage — not used by the reference math
  const float* W1 = (const float*)d_in[2];   // [E, D, F]
  const float* W2 = (const float*)d_in[3];   // [E, F, D]
  float*       Y  = (float*)d_out;           // [E*C, D]

  const size_t xbN  = (size_t)E * C * D;     // bf16 counts
  const size_t w1N  = (size_t)E * D * F;
  const size_t w2N  = (size_t)E * F * D;
  const size_t hN   = (size_t)E * C * F;
  const size_t needFull = (xbN + w1N + w2N + hN) * sizeof(__bf16);  // 448 MB

  dim3 blk(256);

  if (ws_size >= needFull) {
    __bf16* Xb  = (__bf16*)d_ws;
    __bf16* W1t = Xb  + xbN;                 // [E, F, D]
    __bf16* W2t = W1t + w1N;                 // [E, D, F]
    __bf16* H   = W2t + w2N;                 // [E, C, F]

    cvt_bf16_kernel<<<dim3((unsigned)(xbN / 2048)), blk, 0, stream>>>(X, Xb);
    transpose_cvt_kernel<<<dim3(F / 64, D / 64, E), blk, 0, stream>>>(W1, W1t, D, F);
    transpose_cvt_kernel<<<dim3(D / 64, F / 64, E), blk, 0, stream>>>(W2, W2t, F, D);

    moe_gemm<F, D, true, true>
        <<<dim3(F / TILE_N, C / TILE_M, E), blk, 0, stream>>>(Xb, W1t, H, C);
    moe_gemm<D, F, false, false>
        <<<dim3(D / TILE_N, C / TILE_M, E), blk, 0, stream>>>(H, W2t, Y, C);
  } else {
    // Deterministic per-expert fallback: 56 MB of workspace reused per expert.
    const size_t xb1 = (size_t)C * D, w11 = (size_t)D * F, w21 = (size_t)F * D;
    __bf16* Xb  = (__bf16*)d_ws;
    __bf16* W1t = Xb  + xb1;
    __bf16* W2t = W1t + w11;
    __bf16* H   = W2t + w21;
    for (int e = 0; e < E; ++e) {
      const float* Xe  = X  + (size_t)e * C * D;
      const float* W1e = W1 + (size_t)e * D * F;
      const float* W2e = W2 + (size_t)e * F * D;
      float*       Ye  = Y  + (size_t)e * C * D;
      cvt_bf16_kernel<<<dim3((unsigned)(xb1 / 2048)), blk, 0, stream>>>(Xe, Xb);
      transpose_cvt_kernel<<<dim3(F / 64, D / 64, 1), blk, 0, stream>>>(W1e, W1t, D, F);
      transpose_cvt_kernel<<<dim3(D / 64, F / 64, 1), blk, 0, stream>>>(W2e, W2t, F, D);
      moe_gemm<F, D, true, true>
          <<<dim3(F / TILE_N, C / TILE_M, 1), blk, 0, stream>>>(Xb, W1t, H, C);
      moe_gemm<D, F, false, false>
          <<<dim3(D / TILE_N, C / TILE_M, 1), blk, 0, stream>>>(H, W2t, Ye, C);
    }
  }
}
